// ImageEncoder_38010460570114
// MI455X (gfx1250) — compile-verified, matrix operand order checked by reading, not run on hardware
//
#include <hip/hip_runtime.h>
#include <hip/hip_bf16.h>

// ---------------------------------------------------------------------------
// ViT-Base forward for MI455X (gfx1250, wave32, WMMA).
// Core op: v_wmma_f32_16x16x32_bf16, inline int4 dequant on the B operand,
// 2x2 (32x32) register blocking per wave, clamped (branch-free) loads.
// ---------------------------------------------------------------------------

typedef __attribute__((ext_vector_type(16))) __bf16 v16bf;
typedef __attribute__((ext_vector_type(8)))  float  v8f;

#define SEQ   224            // padded sequence (197 valid), 14*16, 7*32
#define NB    8              // batch
#define ROWS  (NB * SEQ)     // 1792 GEMM rows
#define DIM   768
#define NH    12
#define HD    64
#define FFD   3072
#define QKVD  2304
#define NTOK  197
#define NPATCH 196

union Frag {
    v16bf v;
    uint4 q[2];
    __bf16 h[16];
};

__device__ __forceinline__ float gelu_tanh(float x) {
    return 0.5f * x * (1.0f + tanhf(0.7978845608028654f * (x + 0.044715f * x * x * x)));
}

__device__ __forceinline__ int iclamp(int v, int hi) { return v < hi ? v : hi; }

// bf16 fragment load: two 16B global_load_b128, unconditional (row pre-clamped)
__device__ __forceinline__ void load_frag(Frag& f, const __bf16* base, long row,
                                          int ld, int klo) {
    const __bf16* p = base + row * ld + klo;
    f.q[0] = *(const uint4*)p;          // K = klo .. klo+7
    f.q[1] = *(const uint4*)(p + 16);   // K = klo+16 .. klo+23
}

// int4-in-int32 fragment load + dequant -> bf16
__device__ __forceinline__ void load_frag_q(Frag& f, const int* Bq, const float* Bs,
                                            long row, int K, int klo) {
    const int*  pq  = Bq + row * K + klo;
    const float slo = Bs[row * (K >> 6) + (klo >> 6)];
    const float shi = Bs[row * (K >> 6) + ((klo + 16) >> 6)];
    __builtin_prefetch((const void*)(pq + 32), 0, 1);   // next K-chunk of this row
#pragma unroll
    for (int j = 0; j < 8; ++j) {
        f.h[j]     = (__bf16)(((float)pq[j]      - 7.5f) * slo);
        f.h[j + 8] = (__bf16)(((float)pq[j + 16] - 7.5f) * shi);
    }
}

// ---------------------------------------------------------------------------
// Generic batched WMMA GEMM:  C[z][M,N] = act(alpha * A[z] * B[z]^T + bias [+ C])
// A: bf16 [M,lda]; B: bf16 [N,ldb]  OR  quantized int32 [N,K] + scales [N,K/64]
// Each wave computes a 32x32 C block (2x2 WMMA tiles).
// Per-lane fragment layout (16-bit operands, §7.12.2): lane = row;
// lanes<16: elems0-7=K[klo..+7], elems8-15=K[klo+16..+23]; lanes>=16: klo+=8.
// ---------------------------------------------------------------------------
template<int ACT, bool BIAS, bool ACCUM, bool QUANT>
__global__ __launch_bounds__(128) void gemm_wmma(
    float* __restrict__ C,
    const __bf16* __restrict__ A,
    const void*  __restrict__ Bv,
    const float* __restrict__ Bs,
    const float* __restrict__ bias,
    float alpha, int M, int N, int K,
    int lda, int ldb, int ldc,
    long sA, long sB, long sC)
{
    const int lane = threadIdx.x;                        // 0..31 (wave32)
    const int n0   = (blockIdx.x * 4 + threadIdx.y) * 32;
    const int m0   = blockIdx.y * 32;
    if (n0 >= N) return;

    const int z = blockIdx.z;
    A += (long)z * sA;
    C += (long)z * sC;
    const __bf16* Bb = (const __bf16*)Bv + (QUANT ? 0 : (long)z * sB);
    const int*    Bq = (const int*)Bv;                   // quant B never batched

    // clamped rows: OOB rows load valid memory; results dropped by store guard
    const long rA0 = iclamp(m0      + (lane & 15), M - 1);
    const long rA1 = iclamp(m0 + 16 + (lane & 15), M - 1);
    const long rB0 = iclamp(n0      + (lane & 15), N - 1);
    const long rB1 = iclamp(n0 + 16 + (lane & 15), N - 1);
    const int koff = (lane >> 4) << 3;                   // 0 or 8

    v8f a00 = {}, a01 = {}, a10 = {}, a11 = {};
    for (int kk = 0; kk < K; kk += 32) {
        const int klo = kk + koff;
        Frag fa0, fa1, fb0, fb1;
        load_frag(fa0, A, rA0, lda, klo);
        load_frag(fa1, A, rA1, lda, klo);
        if (QUANT) {
            load_frag_q(fb0, Bq, Bs, rB0, K, klo);
            load_frag_q(fb1, Bq, Bs, rB1, K, klo);
        } else {
            load_frag(fb0, Bb, rB0, ldb, klo);
            load_frag(fb1, Bb, rB1, ldb, klo);
        }
        a00 = __builtin_amdgcn_wmma_f32_16x16x32_bf16(false, fa0.v, false, fb0.v, (short)0, a00, false, false);
        a01 = __builtin_amdgcn_wmma_f32_16x16x32_bf16(false, fa0.v, false, fb1.v, (short)0, a01, false, false);
        a10 = __builtin_amdgcn_wmma_f32_16x16x32_bf16(false, fa1.v, false, fb0.v, (short)0, a10, false, false);
        a11 = __builtin_amdgcn_wmma_f32_16x16x32_bf16(false, fa1.v, false, fb1.v, (short)0, a11, false, false);
    }

    // Epilogue. C/D layout: VGPR j -> M = mtile + j + 8*(lane>=16), N = lane&15.
    const int nc   = lane & 15;
    const int mOff = (lane >> 4) << 3;
    const v8f* accs[4] = { &a00, &a01, &a10, &a11 };
#pragma unroll
    for (int ti = 0; ti < 2; ++ti) {
#pragma unroll
        for (int tj = 0; tj < 2; ++tj) {
            const int n = n0 + tj * 16 + nc;
            if (n >= N) continue;
            const float bv = BIAS ? bias[n] : 0.0f;
            const v8f& ac = *accs[ti * 2 + tj];
            const int mBase = m0 + ti * 16 + mOff;
#pragma unroll
            for (int j = 0; j < 8; ++j) {
                const int m = mBase + j;
                if (m < M) {
                    float v = alpha * ac[j] + bv;
                    if (ACCUM) v += C[(long)m * ldc + n];
                    if (ACT == 1) v = gelu_tanh(v);
                    if (ACT == 2) v = v > 0.0f ? v : 0.0f;
                    C[(long)m * ldc + n] = v;
                }
            }
        }
    }
}

// ---------------------------------------------------------------------------
// Elementwise / reduction helper kernels
// ---------------------------------------------------------------------------

__global__ void f32_to_bf16_k(const float* __restrict__ in, __bf16* __restrict__ out, long n) {
    long i = (long)blockIdx.x * blockDim.x + threadIdx.x;
    if (i < n) out[i] = (__bf16)in[i];
}

// img (8,1,3,224,224) -> patch matrix rows [b][t][768] bf16, t=0 & t>=197 zero
__global__ void build_patches_k(const float* __restrict__ img, __bf16* __restrict__ Ap) {
    long i = (long)blockIdx.x * blockDim.x + threadIdx.x;
    if (i >= (long)ROWS * DIM) return;
    const int e = (int)(i % DIM);
    const int t = (int)((i / DIM) % SEQ);
    const int b = (int)(i / ((long)DIM * SEQ));
    float v = 0.0f;
    if (t >= 1 && t <= NPATCH) {
        const int pi = t - 1, py = pi / 14, px = pi % 14;
        const int c = e >> 8, r = (e >> 4) & 15, col = e & 15;
        v = img[(((long)b * 3 + c) * 224 + (py * 16 + r)) * 224 + (px * 16 + col)];
    }
    Ap[i] = (__bf16)v;
}

// after patch GEMM: row0 := cls_token, rows >=197 := 0
__global__ void fixup_x_k(float* __restrict__ x, const float* __restrict__ cls) {
    long i = (long)blockIdx.x * blockDim.x + threadIdx.x;
    if (i >= (long)ROWS * DIM) return;
    const int d = (int)(i % DIM);
    const int t = (int)((i / DIM) % SEQ);
    if (t == 0) x[i] = cls[d];
    else if (t >= NTOK) x[i] = 0.0f;
}

__global__ void residual_k(float* __restrict__ x, const float* __restrict__ o,
                           const float* __restrict__ g) {
    long i = (long)blockIdx.x * blockDim.x + threadIdx.x;
    if (i < (long)ROWS * DIM) x[i] += g[i % DIM] * o[i];
}

// LayerNorm over D=768 per row; 256 threads x 3 elems; writes bf16
__global__ __launch_bounds__(256) void layernorm_k(
    const float* __restrict__ X, const float* __restrict__ g,
    const float* __restrict__ b, __bf16* __restrict__ out)
{
    const int row = blockIdx.x;
    const float* xr = X + (long)row * DIM;
    __bf16* orow = out + (long)row * DIM;
    __shared__ float red[256];
    const int tid = threadIdx.x;

    float a0 = xr[tid], a1 = xr[tid + 256], a2 = xr[tid + 512];
    red[tid] = a0 + a1 + a2;
    __syncthreads();
    for (int o = 128; o > 0; o >>= 1) { if (tid < o) red[tid] += red[tid + o]; __syncthreads(); }
    const float mean = red[0] * (1.0f / DIM);
    __syncthreads();
    const float d0 = a0 - mean, d1 = a1 - mean, d2 = a2 - mean;
    red[tid] = d0 * d0 + d1 * d1 + d2 * d2;
    __syncthreads();
    for (int o = 128; o > 0; o >>= 1) { if (tid < o) red[tid] += red[tid + o]; __syncthreads(); }
    const float rs = rsqrtf(red[0] * (1.0f / DIM) + 1e-6f);
    orow[tid]       = (__bf16)(d0 * rs * g[tid]       + b[tid]);
    orow[tid + 256] = (__bf16)(d1 * rs * g[tid + 256] + b[tid + 256]);
    orow[tid + 512] = (__bf16)(d2 * rs * g[tid + 512] + b[tid + 512]);
}

// qkv [b][t][2304] f32 -> Qh,Kh bf16 [b*12][SEQ][64] (Q pre-scaled 1/8),
// VhT bf16 [b*12][64][SEQ]; padded t rows zeroed.
__global__ void split_heads_k(const float* __restrict__ qkv,
                              __bf16* __restrict__ Qh, __bf16* __restrict__ Kh,
                              __bf16* __restrict__ VT)
{
    long i = (long)blockIdx.x * blockDim.x + threadIdx.x;
    if (i >= (long)ROWS * DIM) return;
    const int d = (int)(i % HD);
    const int h = (int)((i / HD) % NH);
    const int t = (int)((i / DIM) % SEQ);
    const int b = (int)(i / ((long)DIM * SEQ));
    const long src = ((long)b * SEQ + t) * QKVD + h * HD + d;
    const bool vld = t < NTOK;
    const float q = vld ? qkv[src]            : 0.0f;
    const float k = vld ? qkv[src + DIM]      : 0.0f;
    const float v = vld ? qkv[src + 2 * DIM]  : 0.0f;
    const int bh = b * NH + h;
    Qh[((long)bh * SEQ + t) * HD + d] = (__bf16)(q * 0.125f);   // 1/sqrt(64)
    Kh[((long)bh * SEQ + t) * HD + d] = (__bf16)k;
    VT[((long)bh * HD + d) * SEQ + t] = (__bf16)v;
}

// O f32 [b*12][SEQ][64] -> Ocat bf16 [b][SEQ][768]
__global__ void merge_heads_k(const float* __restrict__ O, __bf16* __restrict__ Oc) {
    long i = (long)blockIdx.x * blockDim.x + threadIdx.x;
    if (i >= (long)ROWS * DIM) return;
    const int d = (int)(i % HD);
    const int h = (int)((i / HD) % NH);
    const int t = (int)((i / DIM) % SEQ);
    const int b = (int)(i / ((long)DIM * SEQ));
    Oc[i] = (__bf16)O[(((long)(b * NH + h)) * SEQ + t) * HD + d];
}

// row-softmax over 224 cols, 197 valid; writes bf16 probs (padded cols = 0)
__global__ __launch_bounds__(256) void softmax_k(const float* __restrict__ S,
                                                 __bf16* __restrict__ P)
{
    const long base = ((long)blockIdx.y * SEQ + blockIdx.x) * SEQ;
    const int t = threadIdx.x;
    __shared__ float red[256];
    const float v = (t < NTOK) ? S[base + t] : -3.0e38f;
    red[t] = v;
    __syncthreads();
    for (int o = 128; o > 0; o >>= 1) { if (t < o) red[t] = fmaxf(red[t], red[t + o]); __syncthreads(); }
    const float mx = red[0];
    __syncthreads();
    const float e = (t < NTOK) ? __expf(v - mx) : 0.0f;
    red[t] = e;
    __syncthreads();
    for (int o = 128; o > 0; o >>= 1) { if (t < o) red[t] += red[t + o]; __syncthreads(); }
    const float inv = 1.0f / red[0];
    if (t < SEQ) P[base + t] = (__bf16)(e * inv);
}

// mean over tokens 1..196
__global__ void pool_k(const float* __restrict__ x, float* __restrict__ pooled) {
    const int i = blockIdx.x * blockDim.x + threadIdx.x;
    if (i >= NB * DIM) return;
    const int d = i % DIM, b = i / DIM;
    float s = 0.0f;
    for (int t = 1; t <= NPATCH; ++t) s += x[((long)b * SEQ + t) * DIM + d];
    pooled[i] = s * (1.0f / NPATCH);
}

// ---------------------------------------------------------------------------
// Host side
// ---------------------------------------------------------------------------

template<int ACT, bool BIAS, bool ACCUM, bool QUANT>
static void launch_gemm(hipStream_t st, float* C, const __bf16* A, const void* B,
                        const float* Bs, const float* bias, float alpha,
                        int M, int N, int K, int lda, int ldb, int ldc,
                        long sA, long sB, long sC, int batch)
{
    const int nSuper = (N + 31) / 32;                 // 32-col supertiles
    const int mSuper = (M + 31) / 32;                 // 32-row supertiles
    dim3 grid((nSuper + 3) / 4, mSuper, batch);
    dim3 block(32, 4);
    gemm_wmma<ACT, BIAS, ACCUM, QUANT><<<grid, block, 0, st>>>(
        C, A, B, Bs, bias, alpha, M, N, K, lda, ldb, ldc, sA, sB, sC);
}

static void launch_ew(hipStream_t st, void (*k)(const float*, __bf16*, long),
                      const float* in, __bf16* out, long n)
{
    k<<<dim3((unsigned)((n + 255) / 256)), 256, 0, st>>>(in, out, n);
}

extern "C" void kernel_launch(void* const* d_in, const int* in_sizes, int n_in,
                              void* d_out, int out_size, void* d_ws, size_t ws_size,
                              hipStream_t stream)
{
    (void)in_sizes; (void)n_in; (void)out_size; (void)ws_size;

    const float* img     = (const float*)d_in[0];
    const float* patch_w = (const float*)d_in[1];
    const float* patch_b = (const float*)d_in[2];
    const float* cls_tok = (const float*)d_in[3];
    const float* ln1_g   = (const float*)d_in[4];
    const float* ln1_b   = (const float*)d_in[5];
    const float* ln2_g   = (const float*)d_in[6];
    const float* ln2_b   = (const float*)d_in[7];
    const float* g1      = (const float*)d_in[8];
    const float* g2      = (const float*)d_in[9];
    const int*   qkv_q   = (const int*)  d_in[10];
    const float* qkv_s   = (const float*)d_in[11];
    const float* qkv_b   = (const float*)d_in[12];
    const float* a_qkv   = (const float*)d_in[13];
    const float* b_qkv   = (const float*)d_in[14];
    const int*   proj_q  = (const int*)  d_in[15];
    const float* proj_s  = (const float*)d_in[16];
    const float* proj_b  = (const float*)d_in[17];
    const float* a_proj  = (const float*)d_in[18];
    const float* b_proj  = (const float*)d_in[19];
    const int*   fc1_q   = (const int*)  d_in[20];
    const float* fc1_s   = (const float*)d_in[21];
    const float* fc1_b   = (const float*)d_in[22];
    const int*   fc2_q   = (const int*)  d_in[23];
    const float* fc2_s   = (const float*)d_in[24];
    const float* fc2_b   = (const float*)d_in[25];
    const float* fcn_g   = (const float*)d_in[26];
    const float* fcn_b   = (const float*)d_in[27];
    const float* head_w  = (const float*)d_in[28];
    const float* head_b  = (const float*)d_in[29];
    float* out = (float*)d_out;

    // deterministic workspace carve-up
    char* w = (char*)d_ws;
    size_t off = 0;
    auto carve = [&](size_t bytes) -> char* {
        char* p = w + off;
        off = (off + bytes + 255) & ~(size_t)255;
        return p;
    };
    __bf16* Apatch  = (__bf16*)carve((size_t)ROWS * DIM * 2);
    float*  x       = (float*) carve((size_t)ROWS * DIM * 4);
    __bf16* h       = (__bf16*)carve((size_t)ROWS * DIM * 2);
    float*  qkv     = (float*) carve((size_t)ROWS * QKVD * 4);
    float*  tlora   = (float*) carve((size_t)ROWS * 64 * 4);
    __bf16* tlorab  = (__bf16*)carve((size_t)ROWS * 64 * 2);
    __bf16* Qh      = (__bf16*)carve((size_t)NB * NH * SEQ * HD * 2);
    __bf16* Kh      = (__bf16*)carve((size_t)NB * NH * SEQ * HD * 2);
    __bf16* VT      = (__bf16*)carve((size_t)NB * NH * HD * SEQ * 2);
    float*  Sc      = (float*) carve((size_t)NB * NH * SEQ * SEQ * 4);
    __bf16* Pb      = (__bf16*)carve((size_t)NB * NH * SEQ * SEQ * 2);
    float*  Ob      = (float*) carve((size_t)NB * NH * SEQ * HD * 4);
    __bf16* Ocat    = (__bf16*)carve((size_t)ROWS * DIM * 2);
    float*  obuf    = (float*) carve((size_t)ROWS * DIM * 4);
    float*  ff      = (float*) carve((size_t)ROWS * FFD * 4);
    __bf16* ffb     = (__bf16*)carve((size_t)ROWS * FFD * 2);
    float*  pooled  = (float*) carve((size_t)NB * DIM * 4);
    __bf16* pooledb = (__bf16*)carve((size_t)NB * DIM * 2);
    __bf16* patchWb = (__bf16*)carve((size_t)DIM * DIM * 2);
    __bf16* aQb     = (__bf16*)carve((size_t)12 * 64 * DIM * 2);
    __bf16* bQb     = (__bf16*)carve((size_t)12 * QKVD * 64 * 2);
    __bf16* aPb     = (__bf16*)carve((size_t)12 * 64 * DIM * 2);
    __bf16* bPb     = (__bf16*)carve((size_t)12 * DIM * 64 * 2);
    __bf16* headWb  = (__bf16*)carve((size_t)512 * DIM * 2);

    const long RD = (long)ROWS * DIM;

    // bf16 copies of fp32 weights (small, read many times by WMMA)
    launch_ew(stream, f32_to_bf16_k, patch_w, patchWb, (long)DIM * DIM);
    launch_ew(stream, f32_to_bf16_k, a_qkv,  aQb, (long)12 * 64 * DIM);
    launch_ew(stream, f32_to_bf16_k, b_qkv,  bQb, (long)12 * QKVD * 64);
    launch_ew(stream, f32_to_bf16_k, a_proj, aPb, (long)12 * 64 * DIM);
    launch_ew(stream, f32_to_bf16_k, b_proj, bPb, (long)12 * DIM * 64);
    launch_ew(stream, f32_to_bf16_k, head_w, headWb, (long)512 * DIM);

    // patch embedding
    build_patches_k<<<dim3((unsigned)((RD + 255) / 256)), 256, 0, stream>>>(img, Apatch);
    launch_gemm<0, true, false, false>(stream, x, Apatch, patchWb, nullptr, patch_b,
                                       1.0f, ROWS, DIM, DIM, DIM, DIM, DIM, 0, 0, 0, 1);
    fixup_x_k<<<dim3((unsigned)((RD + 255) / 256)), 256, 0, stream>>>(x, cls_tok);

    const unsigned EWG = (unsigned)((RD + 255) / 256);
    for (int l = 0; l < 12; ++l) {
        const int*   Wq  = qkv_q  + (long)l * QKVD * DIM;
        const float* Wqs = qkv_s  + (long)l * QKVD * (DIM / 64);
        const int*   Wp  = proj_q + (long)l * DIM * DIM;
        const float* Wps = proj_s + (long)l * DIM * (DIM / 64);
        const int*   W1  = fc1_q  + (long)l * FFD * DIM;
        const float* W1s = fc1_s  + (long)l * FFD * (DIM / 64);
        const int*   W2  = fc2_q  + (long)l * DIM * FFD;
        const float* W2s = fc2_s  + (long)l * DIM * (FFD / 64);

        // --- attention half ---
        layernorm_k<<<ROWS, 256, 0, stream>>>(x, ln1_g + l * DIM, ln1_b + l * DIM, h);
        launch_gemm<0, true, false, true>(stream, qkv, h, Wq, Wqs, qkv_b + l * QKVD,
                                          1.0f, ROWS, QKVD, DIM, DIM, DIM, QKVD, 0, 0, 0, 1);
        launch_gemm<0, false, false, false>(stream, tlora, h, aQb + (long)l * 64 * DIM, nullptr,
                                            nullptr, 1.0f, ROWS, 64, DIM, DIM, DIM, 64, 0, 0, 0, 1);
        launch_ew(stream, f32_to_bf16_k, tlora, tlorab, (long)ROWS * 64);
        launch_gemm<0, false, true, false>(stream, qkv, tlorab, bQb + (long)l * QKVD * 64, nullptr,
                                           nullptr, 0.25f, ROWS, QKVD, 64, 64, 64, QKVD, 0, 0, 0, 1);

        split_heads_k<<<EWG, 256, 0, stream>>>(qkv, Qh, Kh, VT);
        // scores = (Q/8) K^T : batched over 96 (b,h)
        launch_gemm<0, false, false, false>(stream, Sc, Qh, Kh, nullptr, nullptr, 1.0f,
                                            SEQ, SEQ, HD, HD, HD, SEQ,
                                            (long)SEQ * HD, (long)SEQ * HD, (long)SEQ * SEQ, NB * NH);
        softmax_k<<<dim3(SEQ, NB * NH), 256, 0, stream>>>(Sc, Pb);
        // O = P V  (B operand is V^T, row-major [64][SEQ])
        launch_gemm<0, false, false, false>(stream, Ob, Pb, VT, nullptr, nullptr, 1.0f,
                                            SEQ, HD, SEQ, SEQ, SEQ, HD,
                                            (long)SEQ * SEQ, (long)HD * SEQ, (long)SEQ * HD, NB * NH);
        merge_heads_k<<<EWG, 256, 0, stream>>>(Ob, Ocat);

        launch_gemm<0, true, false, true>(stream, obuf, Ocat, Wp, Wps, proj_b + l * DIM,
                                          1.0f, ROWS, DIM, DIM, DIM, DIM, DIM, 0, 0, 0, 1);
        launch_gemm<0, false, false, false>(stream, tlora, Ocat, aPb + (long)l * 64 * DIM, nullptr,
                                            nullptr, 1.0f, ROWS, 64, DIM, DIM, DIM, 64, 0, 0, 0, 1);
        launch_ew(stream, f32_to_bf16_k, tlora, tlorab, (long)ROWS * 64);
        launch_gemm<0, false, true, false>(stream, obuf, tlorab, bPb + (long)l * DIM * 64, nullptr,
                                           nullptr, 0.25f, ROWS, DIM, 64, 64, 64, DIM, 0, 0, 0, 1);
        residual_k<<<EWG, 256, 0, stream>>>(x, obuf, g1 + l * DIM);

        // --- MLP half ---
        layernorm_k<<<ROWS, 256, 0, stream>>>(x, ln2_g + l * DIM, ln2_b + l * DIM, h);
        launch_gemm<1, true, false, true>(stream, ff, h, W1, W1s, fc1_b + l * FFD,
                                          1.0f, ROWS, FFD, DIM, DIM, DIM, FFD, 0, 0, 0, 1);
        launch_ew(stream, f32_to_bf16_k, ff, ffb, (long)ROWS * FFD);
        launch_gemm<0, true, false, true>(stream, obuf, ffb, W2, W2s, fc2_b + l * DIM,
                                          1.0f, ROWS, DIM, FFD, FFD, FFD, DIM, 0, 0, 0, 1);
        residual_k<<<EWG, 256, 0, stream>>>(x, obuf, g2 + l * DIM);
    }

    // pool -> LN -> head (+relu) straight into d_out
    pool_k<<<(NB * DIM + 255) / 256, 256, 0, stream>>>(x, pooled);
    layernorm_k<<<NB, 256, 0, stream>>>(pooled, fcn_g, fcn_b, pooledb);
    launch_gemm<2, true, false, false>(stream, out, pooledb, headWb, nullptr, head_b,
                                       1.0f, NB, 512, DIM, DIM, DIM, 512, 0, 0, 0, 1);
}
